// BahdanauAttention_14001593385051
// MI455X (gfx1250) — matrix operand order repaired and checked by reading, not compiled, not run on hardware
//
#include <hip/hip_runtime.h>

typedef __attribute__((ext_vector_type(16))) _Float16 v16h;
typedef __attribute__((ext_vector_type(8)))  _Float16 v8h;
typedef __attribute__((ext_vector_type(4)))  _Float16 v4h;
typedef __attribute__((ext_vector_type(2)))  __fp16   v2fp;
typedef __attribute__((ext_vector_type(4)))  __fp16   v4fp;
typedef __attribute__((ext_vector_type(8)))  float    v8f;

#define B_  64
#define T_  2048
#define D_  512
#define U_  512

// ---------------------------------------------------------------------------
// Branch-free transcendental helpers (prefer native gfx1250 TRANS ops).
// ---------------------------------------------------------------------------
__device__ __forceinline__ float fast_exp(float x) {
#if __has_builtin(__builtin_amdgcn_exp2f)
    return __builtin_amdgcn_exp2f(x * 1.4426950408889634f);   // v_exp_f32
#else
    return __expf(x);
#endif
}

__device__ __forceinline__ float fast_tanh(float x) {
#if __has_builtin(__builtin_amdgcn_tanhf)
    return __builtin_amdgcn_tanhf(x);                          // v_tanh_f32
#elif __has_builtin(__builtin_amdgcn_tanh_f32)
    return __builtin_amdgcn_tanh_f32(x);
#else
    // tanh(x) = (e^{2x}-1)/(e^{2x}+1); clamp keeps e^{2x} finite.
    float xc = fminf(fmaxf(x, -9.0f), 9.0f);
    float t  = fast_exp(2.0f * xc);
#if __has_builtin(__builtin_amdgcn_rcpf)
    return (t - 1.0f) * __builtin_amdgcn_rcpf(t + 1.0f);       // v_rcp_f32
#else
    return (t - 1.0f) / (t + 1.0f);
#endif
#endif
}

// ---------------------------------------------------------------------------
// Kernel 1: W1 [D,U] f32 -> W1T [U,D] f16 (K-major for WMMA B-fragments)
// ---------------------------------------------------------------------------
__global__ void prep_w1t_kernel(const float* __restrict__ W1,
                                _Float16* __restrict__ W1T) {
    int g = blockIdx.x * 256 + threadIdx.x;     // 0 .. 512*512-1
    int u = g >> 9;
    int d = g & 511;
    W1T[(size_t)u * D_ + d] = (_Float16)W1[(size_t)d * U_ + u];
}

// ---------------------------------------------------------------------------
// Kernel 2: pq[b,u] = query[b,:] @ W2[:,u] + b1[u] + b2[u]
// ---------------------------------------------------------------------------
__global__ void proj_q_kernel(const float* __restrict__ query,
                              const float* __restrict__ W2,
                              const float* __restrict__ b1,
                              const float* __restrict__ b2,
                              float* __restrict__ pq) {
    int b = blockIdx.x;
    int u = threadIdx.x;
    float acc = 0.f;
    for (int d = 0; d < D_; ++d)
        acc = fmaf(query[b * D_ + d], W2[(size_t)d * U_ + u], acc);
    pq[b * U_ + u] = acc + b1[u] + b2[u];
}

// ---------------------------------------------------------------------------
// Kernel 3: fused  scores[b,t] = tanh(values[b,t,:]@W1 + pq[b,:]) @ V + bv
// Grid: (T/32, B). Block: 256 threads (8 waves). Each block: 32 T-rows.
// Waves split the 32 N-tiles of U=512 (4 each); each B-fragment feeds TWO
// WMMAs (row tiles 0-15 and 16-31) for 2x W1T reuse.
// ---------------------------------------------------------------------------
__global__ void scores_kernel(const float* __restrict__ values,
                              const _Float16* __restrict__ W1T,
                              const float* __restrict__ pq,
                              const float* __restrict__ Vw,
                              const float* __restrict__ bv,
                              float* __restrict__ scores) {
    __shared__ __align__(32) _Float16 As[32 * D_];   // 32 rows x 512 K f16 = 32KB
    __shared__ float sred[8][32];

    const int b       = blockIdx.y;
    const int rowbase = blockIdx.x * 32;
    const int tid     = threadIdx.x;
    const int wave    = tid >> 5;
    const int lane    = tid & 31;

    // Cooperative load of 32x512 f32 slab -> f16 LDS (contiguous mapping).
    const float4* src = (const float4*)(values + ((size_t)b * T_ + rowbase) * D_);
    #pragma unroll
    for (int it = 0; it < 16; ++it) {
        int idx = it * 256 + tid;                 // 4096 float4 total
        float4 v = src[idx];
#if __has_builtin(__builtin_amdgcn_cvt_pkrtz)
        v2fp lo = __builtin_amdgcn_cvt_pkrtz(v.x, v.y);
        v2fp hi = __builtin_amdgcn_cvt_pkrtz(v.z, v.w);
        v4fp p  = __builtin_shufflevector(lo, hi, 0, 1, 2, 3);
        *(v4h*)&As[idx * 4] = __builtin_bit_cast(v4h, p);
#else
        _Float16* dst = &As[idx * 4];
        dst[0] = (_Float16)v.x; dst[1] = (_Float16)v.y;
        dst[2] = (_Float16)v.z; dst[3] = (_Float16)v.w;
#endif
    }
    __syncthreads();

    // A fragment 16x32, this code's contiguous K order (same as B):
    //   lanes 0-15 : M=lane,   K = k0+0..15
    //   lanes16-31 : M=lane-16,K = k0+16..31
    const int arow = lane & 15;
    const int koff = (lane >> 4) ? 16 : 0;
    // B fragment 32x16, this code's contiguous K order (same as A):
    //   lanes 0-15 : N=lane,   K = k0+0..15   (contiguous in K-major W1T)
    //   lanes16-31 : N=lane-16,K = k0+16..31
    const int n  = lane & 15;
    const int bk = (lane >> 4) ? 16 : 0;

    float s0[8], s1[8];
    #pragma unroll
    for (int i = 0; i < 8; ++i) { s0[i] = 0.f; s1[i] = 0.f; }

    #pragma unroll 1
    for (int jj = 0; jj < 4; ++jj) {
        const int j  = wave + jj * 8;            // N tile 0..31
        const int u0 = j * 16;
        v8f acc0 = {};
        v8f acc1 = {};
        #pragma unroll 4
        for (int ks = 0; ks < 16; ++ks) {
            const int k0 = ks * 32;
            // B fragment: one contiguous 32B global load (L2-resident W1T)
            v16h bfrag = *(const v16h*)&W1T[(size_t)(u0 + n) * D_ + k0 + bk];
            // A fragments for row tiles 0-15 and 16-31 (2 x two 16B LDS loads)
            v8h a0lo = *(const v8h*)&As[arow * D_ + k0 + koff];
            v8h a0hi = *(const v8h*)&As[arow * D_ + k0 + 8 + koff];
            v8h a1lo = *(const v8h*)&As[(16 + arow) * D_ + k0 + koff];
            v8h a1hi = *(const v8h*)&As[(16 + arow) * D_ + k0 + 8 + koff];
            v16h af0 = __builtin_shufflevector(a0lo, a0hi,
                0, 1, 2, 3, 4, 5, 6, 7, 8, 9, 10, 11, 12, 13, 14, 15);
            v16h af1 = __builtin_shufflevector(a1lo, a1hi,
                0, 1, 2, 3, 4, 5, 6, 7, 8, 9, 10, 11, 12, 13, 14, 15);
            acc0 = __builtin_amdgcn_wmma_f32_16x16x32_f16(
                false, af0, false, bfrag, (short)0, acc0, false, false);
            acc1 = __builtin_amdgcn_wmma_f32_16x16x32_f16(
                false, af1, false, bfrag, (short)0, acc1, false, false);
        }
        // C layout: VGPR i, lanes 0-15 -> M=i ; lanes 16-31 -> M=i+8 ; N=lane%16
        const float q  = pq[b * U_ + u0 + n];
        const float vv = Vw[u0 + n];
        #pragma unroll
        for (int i = 0; i < 8; ++i) {
            s0[i] += fast_tanh(acc0[i] + q) * vv;
            s1[i] += fast_tanh(acc1[i] + q) * vv;
        }
    }

    // Reduce over the 16 N-lanes within each half-wave (masks < 16 stay in half)
    #pragma unroll
    for (int m = 8; m >= 1; m >>= 1) {
        #pragma unroll
        for (int i = 0; i < 8; ++i) {
            s0[i] += __shfl_xor(s0[i], m, 32);
            s1[i] += __shfl_xor(s1[i], m, 32);
        }
    }
    if ((lane & 15) == 0) {
        const int rb = (lane >> 4) * 8;
        #pragma unroll
        for (int i = 0; i < 8; ++i) {
            sred[wave][rb + i]      = s0[i];
            sred[wave][16 + rb + i] = s1[i];
        }
    }
    __syncthreads();

    // Sum the 8 waves' partial N-tile contributions, add bv, emit scores.
    if (tid < 32) {
        float tot = 0.f;
        #pragma unroll
        for (int w = 0; w < 8; ++w) tot += sred[w][tid];
        scores[(size_t)b * T_ + rowbase + tid] = tot + bv[0];
    }
}

// ---------------------------------------------------------------------------
// Kernel 4: per-batch softmax over T, write attn, then context = attn^T @ values
// ---------------------------------------------------------------------------
__global__ void softmax_ctx_kernel(const float* __restrict__ values,
                                   const float* __restrict__ scores,
                                   float* __restrict__ out_ctx,
                                   float* __restrict__ out_attn) {
    __shared__ float  sc[T_];          // 8KB
    __shared__ float  red[256];
    __shared__ float4 cred[256];
    const int b   = blockIdx.x;
    const int tid = threadIdx.x;

    float lmax = -3.402823466e38f;
    #pragma unroll
    for (int i = 0; i < 8; ++i) {
        float v = scores[(size_t)b * T_ + tid + i * 256];
        sc[tid + i * 256] = v;
        lmax = fmaxf(lmax, v);
    }
    red[tid] = lmax;
    __syncthreads();
    for (int off = 128; off >= 1; off >>= 1) {
        if (tid < off) red[tid] = fmaxf(red[tid], red[tid + off]);
        __syncthreads();
    }
    const float m = red[0];
    __syncthreads();

    float lsum = 0.f;
    #pragma unroll
    for (int i = 0; i < 8; ++i) {
        float e = fast_exp(sc[tid + i * 256] - m);
        sc[tid + i * 256] = e;
        lsum += e;
    }
    red[tid] = lsum;
    __syncthreads();
    for (int off = 128; off >= 1; off >>= 1) {
        if (tid < off) red[tid] += red[tid + off];
        __syncthreads();
    }
    const float inv = 1.0f / red[0];
    __syncthreads();

    #pragma unroll
    for (int i = 0; i < 8; ++i) {
        int t = tid + i * 256;
        float a = sc[t] * inv;
        sc[t] = a;
        out_attn[(size_t)b * T_ + t] = a;
    }
    __syncthreads();

    // context[b,d] = sum_t attn[t] * values[b,t,d]; float4 columns, 2-way T split
    const int dd = tid & 127;          // float4 column 0..127
    const int tt = tid >> 7;           // 0 or 1
    const float4* v4 = (const float4*)(values + (size_t)b * T_ * D_);
    float4 acc = make_float4(0.f, 0.f, 0.f, 0.f);
    for (int t = tt; t < T_; t += 2) {
        const float a = sc[t];
        float4 x = v4[(size_t)t * (D_ / 4) + dd];
        acc.x += a * x.x; acc.y += a * x.y; acc.z += a * x.z; acc.w += a * x.w;
    }
    cred[tid] = acc;
    __syncthreads();
    if (tid < 128) {
        float4 p = cred[tid];
        float4 q = cred[tid + 128];
        float4 r = make_float4(p.x + q.x, p.y + q.y, p.z + q.z, p.w + q.w);
        ((float4*)(out_ctx + (size_t)b * D_))[dd] = r;
    }
}

// ---------------------------------------------------------------------------
extern "C" void kernel_launch(void* const* d_in, const int* in_sizes, int n_in,
                              void* d_out, int out_size, void* d_ws, size_t ws_size,
                              hipStream_t stream) {
    (void)in_sizes; (void)n_in; (void)out_size; (void)ws_size;

    const float* values = (const float*)d_in[0];
    const float* query  = (const float*)d_in[1];
    const float* W1     = (const float*)d_in[2];
    const float* b1     = (const float*)d_in[3];
    const float* W2     = (const float*)d_in[4];
    const float* b2     = (const float*)d_in[5];
    const float* V      = (const float*)d_in[6];
    const float* bv     = (const float*)d_in[7];

    char* ws = (char*)d_ws;
    _Float16* W1T  = (_Float16*)ws;                                   // 512 KB
    float*    pq   = (float*)(ws + 512 * 1024);                       // 128 KB
    float*    sco  = (float*)(ws + 512 * 1024 + 128 * 1024);          // 512 KB

    float* out_ctx  = (float*)d_out;          // [B, D]
    float* out_attn = out_ctx + B_ * D_;      // [B, T, 1]

    prep_w1t_kernel<<<dim3((D_ * U_) / 256), dim3(256), 0, stream>>>(W1, W1T);
    proj_q_kernel<<<dim3(B_), dim3(U_), 0, stream>>>(query, W2, b1, b2, pq);
    scores_kernel<<<dim3(T_ / 32, B_), dim3(256), 0, stream>>>(
        values, W1T, pq, V, bv, sco);
    softmax_ctx_kernel<<<dim3(B_), dim3(256), 0, stream>>>(
        values, sco, out_ctx, out_attn);
}
